// SVDResidualLinear_78271484003264
// MI455X (gfx1250) — matrix operand order repaired and checked by reading, not compiled, not run on hardware
//
#include <hip/hip_runtime.h>

typedef __attribute__((ext_vector_type(16))) _Float16 v16h;
typedef __attribute__((ext_vector_type(4)))  _Float16 v4h;
typedef __attribute__((ext_vector_type(8)))  float    v8f;
typedef __attribute__((ext_vector_type(4)))  float    v4f;

#define DIM_M 16384   // B*T = 8*2048
#define DIM_N 4096    // OUT
#define DIM_K 4096    // IN
#define RANK  64

// ---------------------------------------------------------------------------
// Kernel 1: convert x (f32 -> f16), vectorized 4-wide.
// ---------------------------------------------------------------------------
__global__ __launch_bounds__(256) void cvt_x_f16(const float* __restrict__ x,
                                                 _Float16* __restrict__ xh) {
  size_t j = ((size_t)blockIdx.x * 256 + threadIdx.x) * 4;
  v4f f = *(const v4f*)(x + j);
  v4h h;
  h.x = (_Float16)f.x; h.y = (_Float16)f.y;
  h.z = (_Float16)f.z; h.w = (_Float16)f.w;
  *(v4h*)(xh + j) = h;
}

// ---------------------------------------------------------------------------
// Kernel 2: Wh[o,i] = f16( weight_main[o,i] + sum_r U[o,r]*S[r]*V[r,i] )
// One block per (1024-column slice, output row o). U[o,:]*S staged in LDS;
// V rows (1 MB total) stay L2-resident across the 4096 o-rows.
// ---------------------------------------------------------------------------
__global__ __launch_bounds__(256) void fuse_weight_f16(
    const float* __restrict__ wm, const float* __restrict__ U,
    const float* __restrict__ S, const float* __restrict__ V,
    _Float16* __restrict__ Wh) {
  __shared__ float us[RANK];
  const int o = blockIdx.y;
  const int t = threadIdx.x;
  if (t < RANK) us[t] = U[(size_t)o * RANK + t] * S[t];
  __syncthreads();
  const int i0 = blockIdx.x * 1024 + t * 4;
  v4f acc = *(const v4f*)(wm + (size_t)o * DIM_K + i0);
#pragma unroll 8
  for (int r = 0; r < RANK; ++r) {
    v4f v = *(const v4f*)(V + (size_t)r * DIM_K + i0);
    acc += us[r] * v;
  }
  v4h h;
  h.x = (_Float16)acc.x; h.y = (_Float16)acc.y;
  h.z = (_Float16)acc.z; h.w = (_Float16)acc.w;
  *(v4h*)(Wh + (size_t)o * DIM_K + i0) = h;
}

// ---------------------------------------------------------------------------
// Kernel 3: out[m,n] = sum_k A[m,k]*Wh[n,k] + bias[n]
// 256 threads = 8 waves; block tile 128x128; wave tile 32x64 (2x4 WMMA accs).
// Fragments loaded directly from global (weight is L2-resident, 32 MB << 192 MB)
// using the exact CDNA5 VGPR layouts for 16-bit A (16x32) and B (32x16).
// ---------------------------------------------------------------------------
__global__ __launch_bounds__(256) void wmma_gemm_f16(
    const _Float16* __restrict__ A, const _Float16* __restrict__ Bw,
    const float* __restrict__ bias, float* __restrict__ out) {
  const int lane = threadIdx.x & 31;
  const int wave = threadIdx.x >> 5;
  const int hi   = lane >> 4;   // which 16-lane half
  const int l15  = lane & 15;

  const int m0 = blockIdx.y * 128 + (wave & 3) * 32;  // wave rows m0..m0+31
  const int n0 = blockIdx.x * 128 + (wave >> 2) * 64; // wave cols n0..n0+63

  // A fragment (16x32 f16): lane<16 holds K 0..7 & 16..23 of row m0+l15;
  // lane>=16 holds K 8..15 & 24..31  ->  base shifted by hi*8 halves.
  const _Float16* aP0 = A + (size_t)(m0 + l15) * DIM_K + hi * 8;
  const _Float16* aP1 = aP0 + (size_t)16 * DIM_K;

  // B fragment (32x16 f16): lane holds column N = n-row of W, K 0..7 & 16..23
  // (lane<16) or K 8..15 & 24..31 (lane>=16)  ->  base shifted by hi*8 halves.
  const _Float16* bP[4];
#pragma unroll
  for (int nt = 0; nt < 4; ++nt)
    bP[nt] = Bw + (size_t)(n0 + nt * 16 + l15) * DIM_K + hi * 8;

  v8f acc[2][4] = {};

  for (int k = 0; k < DIM_K; k += 32) {
    union Frag { v16h h; v4f f[2]; };
    Frag a0, a1, b[4];
    a0.f[0] = *(const v4f*)(aP0 + k);
    a0.f[1] = *(const v4f*)(aP0 + k + 16);
    a1.f[0] = *(const v4f*)(aP1 + k);
    a1.f[1] = *(const v4f*)(aP1 + k + 16);
#pragma unroll
    for (int nt = 0; nt < 4; ++nt) {
      b[nt].f[0] = *(const v4f*)(bP[nt] + k);
      b[nt].f[1] = *(const v4f*)(bP[nt] + k + 16);
    }
#pragma unroll
    for (int nt = 0; nt < 4; ++nt) {
      acc[0][nt] = __builtin_amdgcn_wmma_f32_16x16x32_f16(
          false, a0.h, false, b[nt].h, (short)0, acc[0][nt], false, false);
      acc[1][nt] = __builtin_amdgcn_wmma_f32_16x16x32_f16(
          false, a1.h, false, b[nt].h, (short)0, acc[1][nt], false, false);
    }
  }

  // C/D layout: VGPR r -> row (r + hi*8), col l15.
#pragma unroll
  for (int nt = 0; nt < 4; ++nt) {
    const float bv = bias[n0 + nt * 16 + l15];
#pragma unroll
    for (int mt = 0; mt < 2; ++mt) {
      float* o = out + (size_t)(m0 + mt * 16 + hi * 8) * DIM_N + n0 + nt * 16 + l15;
#pragma unroll
      for (int r = 0; r < 8; ++r)
        o[(size_t)r * DIM_N] = acc[mt][nt][r] + bv;
    }
  }
}

// ---------------------------------------------------------------------------
extern "C" void kernel_launch(void* const* d_in, const int* in_sizes, int n_in,
                              void* d_out, int out_size, void* d_ws, size_t ws_size,
                              hipStream_t stream) {
  (void)in_sizes; (void)n_in; (void)out_size; (void)ws_size;
  const float* x    = (const float*)d_in[0];
  const float* wm   = (const float*)d_in[1];
  const float* U    = (const float*)d_in[2];
  const float* S    = (const float*)d_in[3];
  const float* V    = (const float*)d_in[4];
  const float* bias = (const float*)d_in[5];
  float* out = (float*)d_out;

  // workspace: xh (128 MB) | Wh (32 MB)
  _Float16* xh = (_Float16*)d_ws;
  _Float16* Wh = xh + (size_t)DIM_M * DIM_K;

  cvt_x_f16<<<((size_t)DIM_M * DIM_K) / (256 * 4), 256, 0, stream>>>(x, xh);
  fuse_weight_f16<<<dim3(DIM_K / 1024, DIM_N), 256, 0, stream>>>(wm, U, S, V, Wh);
  wmma_gemm_f16<<<dim3(DIM_N / 128, DIM_M / 128), 256, 0, stream>>>(xh, Wh, bias, out);
}
